// fftSinkhornLinear_74371653697769
// MI455X (gfx1250) — matrix operand order repaired and checked by reading, not compiled, not run on hardware
//
#include <hip/hip_runtime.h>
#include <cstdint>

typedef __attribute__((ext_vector_type(16))) _Float16 v16h;
typedef __attribute__((ext_vector_type(8)))  float    v8f;

#define BROWS 16384
#define DDIM  4096
#define USE_ASYNC 1

// ---------------------------------------------------------------------------
// Kernel 1: per-row 4096-pt in-place DIT FFT in LDS, |.|, row-max normalize,
// write f16 A-matrix for the WMMA GEMM.
// ---------------------------------------------------------------------------
__global__ __launch_bounds__(256) void fft_rowmag_kernel(
    const float* __restrict__ x, _Float16* __restrict__ A16) {
  __shared__ float re[DDIM];
  __shared__ float im[DDIM];
  __shared__ float red[256];

  const int row = blockIdx.x;
  const int t   = threadIdx.x;
  const float* __restrict__ xr = x + (size_t)row * DDIM;

  for (int k = 0; k < 16; ++k) {
    int c  = t + 256 * k;
    int rc = (int)(__brev((unsigned)c) >> 20);  // 12-bit reversal
    re[rc] = xr[c];
    im[rc] = 0.0f;
  }
  __syncthreads();

  for (int stage = 1; stage <= 12; ++stage) {
    const int half = 1 << (stage - 1);
    for (int b8 = 0; b8 < 8; ++b8) {
      int b   = t + 256 * b8;
      int pos = b & (half - 1);
      int grp = b >> (stage - 1);
      int i0  = (grp << stage) + pos;
      int i1  = i0 + half;
      float ang = -3.14159265358979323846f * (float)pos / (float)half;
      float sw, cw;
      __sincosf(ang, &sw, &cw);
      float ur = re[i0], ui = im[i0];
      float xr1 = re[i1], xi1 = im[i1];
      float vr = cw * xr1 - sw * xi1;
      float vi = cw * xi1 + sw * xr1;
      re[i0] = ur + vr;  im[i0] = ui + vi;
      re[i1] = ur - vr;  im[i1] = ui - vi;
    }
    __syncthreads();
  }

  float mag[16];
  float lmax = 0.0f;
  for (int k = 0; k < 16; ++k) {
    int c = t + 256 * k;
    float m = sqrtf(re[c] * re[c] + im[c] * im[c]);
    mag[k] = m;
    lmax = fmaxf(lmax, m);
  }
  red[t] = lmax;
  __syncthreads();
  for (int o = 128; o > 0; o >>= 1) {
    if (t < o) red[t] = fmaxf(red[t], red[t + o]);
    __syncthreads();
  }
  const float inv = 1.0f / (red[0] + 1e-6f);

  _Float16* __restrict__ ar = A16 + (size_t)row * DDIM;
  for (int k = 0; k < 16; ++k) {
    int c = t + 256 * k;
    ar[c] = (_Float16)(mag[k] * inv);
  }
}

// ---------------------------------------------------------------------------
// Sinkhorn: row pass (optionally fusing the pending column subtraction).
// ---------------------------------------------------------------------------
__global__ __launch_bounds__(256) void row_lse_sub_kernel(
    const float* __restrict__ src, float* __restrict__ dst,
    const float* __restrict__ colv) {
  __shared__ float sm[256];
  __shared__ float ss[256];
  const int row = blockIdx.x;
  const int t   = threadIdx.x;
  const float* __restrict__ p = src + (size_t)row * DDIM;

  float v[16];
  float m = -INFINITY, s = 0.0f;
  for (int k = 0; k < 16; ++k) {
    int c = t + 256 * k;
    float xv = p[c];
    if (colv) xv -= colv[c];      // lazy column normalization from prev iter
    v[k] = xv;
    if (xv > m) { s = s * expf(m - xv) + 1.0f; m = xv; }
    else        { s += expf(xv - m); }
  }
  sm[t] = m; ss[t] = s;
  __syncthreads();
  for (int o = 128; o > 0; o >>= 1) {
    if (t < o) {
      float m2 = sm[t + o], s2 = ss[t + o];
      float M  = fmaxf(sm[t], m2);
      ss[t] = ss[t] * expf(sm[t] - M) + s2 * expf(m2 - M);
      sm[t] = M;
    }
    __syncthreads();
  }
  const float lse = sm[0] + logf(ss[0]);
  float* __restrict__ q = dst + (size_t)row * DDIM;
  for (int k = 0; k < 16; ++k) q[t + 256 * k] = v[k] - lse;
}

// Column LSE, split over 16 row-chunks for parallelism, then combined.
__global__ __launch_bounds__(256) void col_lse_part_kernel(
    const float* __restrict__ P, float* __restrict__ pm, float* __restrict__ ps) {
  const int c   = blockIdx.x * 256 + threadIdx.x;   // 0..4095
  const int rch = blockIdx.y;                       // 0..15
  float m = -INFINITY, s = 0.0f;
  for (int r = rch * 256; r < rch * 256 + 256; ++r) {
    float xv = P[(size_t)r * DDIM + c];
    if (xv > m) { s = s * expf(m - xv) + 1.0f; m = xv; }
    else        { s += expf(xv - m); }
  }
  pm[(size_t)rch * DDIM + c] = m;
  ps[(size_t)rch * DDIM + c] = s;
}

__global__ __launch_bounds__(256) void col_lse_comb_kernel(
    const float* __restrict__ pm, const float* __restrict__ ps,
    float* __restrict__ colv) {
  const int c = blockIdx.x * 256 + threadIdx.x;
  float m = -INFINITY, s = 0.0f;
  for (int i = 0; i < 16; ++i) {
    float m2 = pm[(size_t)i * DDIM + c], s2 = ps[(size_t)i * DDIM + c];
    float M  = fmaxf(m, m2);
    s = s * expf(m - M) + s2 * expf(m2 - M);
    m = M;
  }
  colv[c] = m + logf(s);
}

// Final: subtract col-LSE, exponentiate -> P (f32 output) and transposed f16
// copy Bt[n][k] via LDS-tiled 128x128 transpose (coalesced both directions).
__global__ __launch_bounds__(256) void col_sub_exp_tr_kernel(
    float* __restrict__ P, const float* __restrict__ colv,
    _Float16* __restrict__ Bt) {
  __shared__ _Float16 sT[128 * 136];
  const int r0 = blockIdx.y * 128;
  const int c0 = blockIdx.x * 128;
  const int t  = threadIdx.x;
  for (int i = 0; i < 64; ++i) {
    int idx = t + 256 * i;
    int rl = idx >> 7, cl = idx & 127;
    size_t gi = (size_t)(r0 + rl) * DDIM + c0 + cl;
    float pv = expf(P[gi] - colv[c0 + cl]);
    P[gi] = pv;
    sT[cl * 136 + rl] = (_Float16)pv;
  }
  __syncthreads();
  for (int i = 0; i < 64; ++i) {
    int idx = t + 256 * i;
    int cl = idx >> 7, rl = idx & 127;
    Bt[(size_t)(c0 + cl) * DDIM + r0 + rl] = sT[cl * 136 + rl];
  }
}

// ---------------------------------------------------------------------------
// WMMA GEMM: C[16384x4096] = A(f16) * P, B supplied transposed (Bt[n][k]).
// Block = 256 threads = 8 waves. Block tile 256(M) x 64(N).
// Wave tile 32(M) x 64(N) = 8 accumulators. K chunked by 128 through a
// double-buffered LDS B tile filled with async global->LDS copies.
// ---------------------------------------------------------------------------
union FragA {
  uint4 u[2];
  v16h  h;
};

__device__ __forceinline__ void async_copy_b128(_Float16* l, const _Float16* g) {
#if USE_ASYNC
  unsigned lo = (unsigned)(uintptr_t)l;
  unsigned long long ga = (unsigned long long)(uintptr_t)g;
  asm volatile("global_load_async_to_lds_b128 %0, %1, off"
               :: "v"(lo), "v"(ga) : "memory");
#else
  *(uint4*)l = *(const uint4*)g;
#endif
}

__device__ __forceinline__ void wait_async_all() {
#if USE_ASYNC
  asm volatile("s_wait_asynccnt 0x0" ::: "memory");
#endif
}

// Copy one 128(K) x 64(N) B chunk into LDS (row stride 136 halves).
__device__ __forceinline__ void load_btile(const _Float16* __restrict__ Bt,
                                           int n0, int kc,
                                           _Float16* __restrict__ sbuf, int t) {
  const int n   = t >> 2;       // 0..63
  const int seg = t & 3;        // 0..3, 64B each
  const _Float16* __restrict__ g = Bt + (size_t)(n0 + n) * DDIM + kc + seg * 32;
  _Float16* __restrict__ l = sbuf + n * 136 + seg * 32;
  async_copy_b128(l,      g);
  async_copy_b128(l + 8,  g + 8);
  async_copy_b128(l + 16, g + 16);
  async_copy_b128(l + 24, g + 24);
}

__device__ __forceinline__ void load_afrag(FragA& f, const _Float16* arow,
                                           int k, int h) {
  f.u[0] = *(const uint4*)(arow + k + 8 * h);
  f.u[1] = *(const uint4*)(arow + k + 16 + 8 * h);
}

__device__ __forceinline__ v8f wmma_f16(v16h a, v16h b, v8f c) {
  return __builtin_amdgcn_wmma_f32_16x16x32_f16(false, a, false, b, (short)0, c,
                                                false, false);
}

__global__ __launch_bounds__(256) void wmma_gemm_kernel(
    const _Float16* __restrict__ A, const _Float16* __restrict__ Bt,
    float* __restrict__ C) {
  __shared__ _Float16 sB[2][64 * 136];

  const int t    = threadIdx.x;
  const int lane = t & 31;
  const int wave = t >> 5;            // 0..7
  const int h    = lane >> 4;
  const int l16  = lane & 15;

  const int m0 = blockIdx.x * 256 + wave * 32;
  const int n0 = blockIdx.y * 64;

  const _Float16* __restrict__ arow0 = A + (size_t)(m0 + l16) * DDIM;
  const _Float16* __restrict__ arow1 = arow0 + (size_t)16 * DDIM;

  v8f acc[8] = {};

  load_btile(Bt, n0, 0, sB[0], t);
  wait_async_all();
  __syncthreads();

  for (int c = 0; c < DDIM / 128; ++c) {
    const int kc = c * 128;
    if (c + 1 < DDIM / 128) load_btile(Bt, n0, kc + 128, sB[(c + 1) & 1], t);

    const _Float16* __restrict__ sb = sB[c & 1];
    __builtin_prefetch(arow0 + kc + 256, 0, 3);
    __builtin_prefetch(arow1 + kc + 256, 0, 3);

    FragA a0, a1, na0, na1;
    load_afrag(a0, arow0, kc, h);
    load_afrag(a1, arow1, kc, h);

#pragma unroll
    for (int ks = 0; ks < 4; ++ks) {
      const int kl = ks * 32;
      if (ks < 3) {                       // pipeline next K-step's A fragments
        load_afrag(na0, arow0, kc + kl + 32, h);
        load_afrag(na1, arow1, kc + kl + 32, h);
      }
#pragma unroll
      for (int nt = 0; nt < 4; ++nt) {
        FragA fb;
        const _Float16* bp = sb + (nt * 16 + l16) * 136 + kl + 8 * h;
        fb.u[0] = *(const uint4*)(bp);
        fb.u[1] = *(const uint4*)(bp + 16);
        acc[nt]     = wmma_f16(a0.h, fb.h, acc[nt]);
        acc[4 + nt] = wmma_f16(a1.h, fb.h, acc[4 + nt]);
      }
      a0 = na0;
      a1 = na1;
    }

    wait_async_all();
    __syncthreads();
  }

  // C layout: acc element i -> row M = 8h + i, col N = l16 (+16 per nt).
#pragma unroll
  for (int mt = 0; mt < 2; ++mt) {
    float* __restrict__ cr = C + (size_t)(m0 + mt * 16 + 8 * h) * DDIM + n0 + l16;
#pragma unroll
    for (int i = 0; i < 8; ++i) {
#pragma unroll
      for (int nt = 0; nt < 4; ++nt) {
        cr[(size_t)i * DDIM + nt * 16] = acc[mt * 4 + nt][i];
      }
    }
  }
}

// ---------------------------------------------------------------------------
extern "C" void kernel_launch(void* const* d_in, const int* in_sizes, int n_in,
                              void* d_out, int out_size, void* d_ws, size_t ws_size,
                              hipStream_t stream) {
  const float* x  = (const float*)d_in[0];        // 16384 x 4096
  const float* lw = (const float*)d_in[1];        // 4096 x 4096

  float* outC = (float*)d_out;                    // xfft @ P
  float* outP = outC + (size_t)BROWS * DDIM;      // P (also logP scratch)

  char* ws = (char*)d_ws;
  _Float16* A16  = (_Float16*)ws;                                  // 128 MB
  ws += (size_t)BROWS * DDIM * 2;
  _Float16* Bt16 = (_Float16*)ws;                                  //  32 MB
  ws += (size_t)DDIM * DDIM * 2;
  float* colv = (float*)ws;  ws += (size_t)DDIM * 4;               //  16 KB
  float* pm   = (float*)ws;  ws += (size_t)16 * DDIM * 4;          // 256 KB
  float* ps   = (float*)ws;                                        // 256 KB

  // 1) FFT magnitude + row-max normalize -> f16 A matrix.
  fft_rowmag_kernel<<<BROWS, 256, 0, stream>>>(x, A16);

  // 2) Sinkhorn, 5 iterations. Column subtraction of iteration i is fused
  //    into iteration i+1's row pass; the last one into the exp kernel.
  for (int it = 0; it < 5; ++it) {
    const float* src = (it == 0) ? lw : outP;
    const float* cv  = (it == 0) ? (const float*)nullptr : colv;
    row_lse_sub_kernel<<<DDIM, 256, 0, stream>>>(src, outP, cv);
    col_lse_part_kernel<<<dim3(DDIM / 256, 16), 256, 0, stream>>>(outP, pm, ps);
    col_lse_comb_kernel<<<DDIM / 256, 256, 0, stream>>>(pm, ps, colv);
  }
  col_sub_exp_tr_kernel<<<dim3(DDIM / 128, DDIM / 128), 256, 0, stream>>>(
      outP, colv, Bt16);

  // 3) WMMA GEMM: outC = A16 * P (B consumed as transposed f16 copy in LDS).
  dim3 grid(BROWS / 256, DDIM / 64);
  wmma_gemm_kernel<<<grid, 256, 0, stream>>>(A16, Bt16, outC);
}
